// SelfAttention_12317966205034
// MI455X (gfx1250) — matrix operand order repaired and checked by reading, not compiled, not run on hardware
//
#include <hip/hip_runtime.h>
#include <hip/hip_bf16.h>
#include <stdint.h>

// ---------------------------------------------------------------------------
// Self-attention (B=2, S=4096, D=512, H=8, dh=64), all matmuls on
// v_wmma_f32_16x16x32_bf16, async global->LDS staging, double-buffered.
// Pipeline: prep(x->bf16, W^T->bf16) ; GEMM1 ; V-transpose ; flash-attn ; GEMM2
// ---------------------------------------------------------------------------

typedef __bf16 bf16_t;
typedef __attribute__((ext_vector_type(16))) __bf16 v16bf;
typedef __attribute__((ext_vector_type(8)))  __bf16 v8bf;
typedef __attribute__((ext_vector_type(8)))  float  v8f;
typedef __attribute__((ext_vector_type(4)))  int    v4i;

static __device__ __forceinline__ bf16_t f2bf(float f) {
  unsigned int u = __builtin_bit_cast(unsigned int, f);
  u = (u + 0x7FFFu + ((u >> 16) & 1u)) >> 16;
  unsigned short s = (unsigned short)u;
  return __builtin_bit_cast(bf16_t, s);
}
static __device__ __forceinline__ void store_out(float*  p, float v) { *p = v; }
static __device__ __forceinline__ void store_out(bf16_t* p, float v) { *p = f2bf(v); }

union bfrag { v16bf v; v8bf h8[2]; };

// 16-byte global -> LDS copy; async (ASYNCcnt) on gfx1250 when available.
static __device__ __forceinline__ void cp16(void* lds, const void* g) {
#if defined(__gfx1250__) && __has_builtin(__builtin_amdgcn_global_load_async_to_lds_b128)
  __builtin_amdgcn_global_load_async_to_lds_b128(
      (__attribute__((address_space(1))) v4i*)g,
      (__attribute__((address_space(3))) v4i*)lds, 0, 0);
#else
  *(v8bf*)lds = *(const v8bf*)g;
#endif
}
static __device__ __forceinline__ void async_wait0() {
#if defined(__gfx1250__) && __has_builtin(__builtin_amdgcn_s_wait_asynccnt)
  __builtin_amdgcn_s_wait_asynccnt(0);
#endif
}

static __device__ __forceinline__ v8f wmma_bf16(const bfrag& a, const bfrag& b, v8f c) {
  return __builtin_amdgcn_wmma_f32_16x16x32_bf16(false, a.v, false, b.v,
                                                 (short)0, c, false, false);
}

// max-reduce across the 16-lane group (lanes sharing a C-fragment row).
// Prefers v_permlane16_b32 (VALU pipe) over ds_bpermute (LDS pipe).
static __device__ __forceinline__ float xor16_max(float v) {
#if __has_builtin(__builtin_amdgcn_permlane16)
  // nibble selectors: lane j reads lane j^m within its 16-lane row
  const unsigned lo[4] = {0x67452301u, 0x54761032u, 0x32107654u, 0xFEDCBA98u};
  const unsigned hi[4] = {0xEFCDAB89u, 0xDCFE98BAu, 0xBA98FEDCu, 0x76543210u};
#pragma unroll
  for (int s = 0; s < 4; ++s) {
    unsigned u = __builtin_bit_cast(unsigned, v);
    unsigned p = __builtin_amdgcn_permlane16(u, u, lo[s], hi[s], false, false);
    v = fmaxf(v, __builtin_bit_cast(float, p));
  }
#else
#pragma unroll
  for (int off = 1; off < 16; off <<= 1)
    v = fmaxf(v, __shfl_xor(v, off, 32));
#endif
  return v;
}

// ---------------------------------------------------------------------------
// Prep: elementwise f32 -> bf16
// ---------------------------------------------------------------------------
__global__ __launch_bounds__(256)
void cvt_bf16_kernel(const float* __restrict__ src, bf16_t* __restrict__ dst, int n) {
  const int i = (blockIdx.x * 256 + threadIdx.x) * 8;
  if (i + 8 > n) return;
  const float4 a = *(const float4*)(src + i);
  const float4 b = *(const float4*)(src + i + 4);
  v8bf o;
  o[0] = f2bf(a.x); o[1] = f2bf(a.y); o[2] = f2bf(a.z); o[3] = f2bf(a.w);
  o[4] = f2bf(b.x); o[5] = f2bf(b.y); o[6] = f2bf(b.z); o[7] = f2bf(b.w);
  *(v8bf*)(dst + i) = o;
}

// Prep: transpose+convert f32 [R][C] -> bf16 [C][R], 64x64 tiles through LDS.
__global__ __launch_bounds__(256)
void transpose_cvt_kernel(const float* __restrict__ src, bf16_t* __restrict__ dst,
                          int R, int C) {
  __shared__ __align__(16) bf16_t T[64][80];
  const int c0 = blockIdx.x * 64, r0 = blockIdx.y * 64;
  const int rr = threadIdx.x >> 2, seg = (threadIdx.x & 3) * 16;
  {
    const float* sp = src + (size_t)(r0 + rr) * C + c0 + seg;
#pragma unroll
    for (int j = 0; j < 16; ++j) T[rr][seg + j] = f2bf(sp[j]);
  }
  __syncthreads();
  {
    const int cc = rr;  // column of src handled on write side
    bf16_t* dp = dst + (size_t)(c0 + cc) * R + r0 + seg;
    v8bf o0, o1;
#pragma unroll
    for (int j = 0; j < 8; ++j) { o0[j] = T[seg + j][cc]; o1[j] = T[seg + 8 + j][cc]; }
    *(v8bf*)(dp)     = o0;
    *(v8bf*)(dp + 8) = o1;
  }
}

// Prep: V section of qkv (bf16) -> Vt[b][h][dh][S] (bf16), 64x64 tiles.
__global__ __launch_bounds__(256)
void v_transpose_kernel(const bf16_t* __restrict__ qkv, bf16_t* __restrict__ Vt) {
  constexpr int S = 4096, TD = 1536;
  __shared__ __align__(16) bf16_t T[64][80];
  const int s0 = blockIdx.x * 64, h = blockIdx.y, b = blockIdx.z;
  const int rr = threadIdx.x >> 2, seg = (threadIdx.x & 3) * 16;
  {
    const bf16_t* sp = qkv + (size_t)(b * S + s0 + rr) * TD + 1024 + h * 64 + seg;
    *(v8bf*)&T[rr][seg]     = *(const v8bf*)(sp);
    *(v8bf*)&T[rr][seg + 8] = *(const v8bf*)(sp + 8);
  }
  __syncthreads();
  {
    const int dd = rr;  // dh index on write side
    bf16_t* dp = Vt + ((size_t)(b * 8 + h) * 64 + dd) * S + s0 + seg;
    v8bf o0, o1;
#pragma unroll
    for (int j = 0; j < 8; ++j) { o0[j] = T[seg + j][dd]; o1[j] = T[seg + 8 + j][dd]; }
    *(v8bf*)(dp)     = o0;
    *(v8bf*)(dp + 8) = o1;
  }
}

// ---------------------------------------------------------------------------
// GEMM + bias: C[M,N] = A[M,K](bf16) @ Bt[N,K]^T(bf16) + bias[N]
// 256 thr, block tile 128x128, wave = 32 rows x 64 cols (2 A-frags reuse
// 4 B-frags -> 8 wmma / 12 ds_load_b128 per K-step), double-buffered async.
// ---------------------------------------------------------------------------
template <typename OT>
__global__ __launch_bounds__(256)
void gemm_bias_kernel(const bf16_t* __restrict__ A, const bf16_t* __restrict__ Bt,
                      const float* __restrict__ bias, OT* __restrict__ C,
                      int M, int N, int K)
{
  __shared__ __align__(16) bf16_t Ash[2][128][32];
  __shared__ __align__(16) bf16_t Bsh[2][128][32];

  const int tid  = threadIdx.x;
  const int wave = tid >> 5, lane = tid & 31;
  const int half = lane >> 4, l16 = lane & 15;
  const int wm = wave >> 1, wn = wave & 1;
  const int m0 = blockIdx.x * 128, n0 = blockIdx.y * 128;

  const int la_row = tid >> 1, la_seg = (tid & 1) * 16;  // 128x32 / 256 = 16 each
  const int lb_n   = tid >> 1, lb_seg = (tid & 1) * 16;  // 128x32 / 256 = 16 each

  const int nsteps = K >> 5;

  auto issue = [&](int kidx, int buf) {
    const bf16_t* as = A + (size_t)(m0 + la_row) * K + (kidx << 5) + la_seg;
    cp16(&Ash[buf][la_row][la_seg], as);
    cp16(&Ash[buf][la_row][la_seg + 8], as + 8);
    const bf16_t* bs = Bt + (size_t)(n0 + lb_n) * K + (kidx << 5) + lb_seg;
    cp16(&Bsh[buf][lb_n][lb_seg], bs);
    cp16(&Bsh[buf][lb_n][lb_seg + 8], bs + 8);
  };

  v8f acc[2][4] = {};
  issue(0, 0);

  for (int kidx = 0; kidx < nsteps; ++kidx) {
    const int cur = kidx & 1;
    async_wait0();
    __syncthreads();
    if (kidx + 1 < nsteps) issue(kidx + 1, cur ^ 1);

    bfrag af[2];
#pragma unroll
    for (int ms = 0; ms < 2; ++ms) {
      const bf16_t* ap = &Ash[cur][wm * 32 + ms * 16 + l16][half * 8];
      af[ms].h8[0] = *(const v8bf*)(ap);
      af[ms].h8[1] = *(const v8bf*)(ap + 16);
    }
#pragma unroll
    for (int nt = 0; nt < 4; ++nt) {
      bfrag bf;
      const bf16_t* bp = &Bsh[cur][wn * 64 + nt * 16 + l16][half * 8];
      bf.h8[0] = *(const v8bf*)(bp);
      bf.h8[1] = *(const v8bf*)(bp + 16);
#pragma unroll
      for (int ms = 0; ms < 2; ++ms)
        acc[ms][nt] = wmma_bf16(af[ms], bf, acc[ms][nt]);
    }
  }

#pragma unroll
  for (int ms = 0; ms < 2; ++ms)
#pragma unroll
    for (int nt = 0; nt < 4; ++nt) {
      const int col = n0 + wn * 64 + nt * 16 + l16;
      const float bv = bias[col];
#pragma unroll
      for (int r = 0; r < 8; ++r) {
        const int row = m0 + wm * 32 + ms * 16 + r + half * 8;
        store_out(&C[(size_t)row * N + col], acc[ms][nt][r] + bv);
      }
    }
}

// ---------------------------------------------------------------------------
// Flash attention. qkv bf16 [B*S,1536] (Q|K|V, head h at cols h*64),
// Vt bf16 [B][H][64][S]. Block = 128 queries x one (b,h); wave = 16 queries.
// Double-buffered async K/V tiles of 64 keys. Row sums via P @ ones WMMA.
// ---------------------------------------------------------------------------
__global__ __launch_bounds__(256)
void flash_attn_kernel(const bf16_t* __restrict__ qkv, const bf16_t* __restrict__ Vt,
                       bf16_t* __restrict__ Oout, const int* __restrict__ causal_flag)
{
  constexpr int S = 4096, TD = 1536, D = 512, DH = 64;

  __shared__ __align__(16) bf16_t Ksh[2][64][64];    // [k][d]
  __shared__ __align__(16) bf16_t Vsh[2][64][64];    // [d][k]
  __shared__ __align__(16) bf16_t Psh[8][16][64];    // per-wave P scratch

  const int tid  = threadIdx.x;
  const int wave = tid >> 5, lane = tid & 31;
  const int half = lane >> 4, l16 = lane & 15;

  const int q0 = blockIdx.x * 128;
  const int h  = blockIdx.y;
  const int b  = blockIdx.z;
  const int qw = q0 + wave * 16;
  const bool causal = (causal_flag[0] != 0);

  // Q fragments, kept in registers for the whole pass (2 K-steps over dh)
  bfrag qf[2];
  {
    const bf16_t* qrow = qkv + (size_t)(b * S + qw + l16) * TD + h * DH;
#pragma unroll
    for (int ks = 0; ks < 2; ++ks) {
      qf[ks].h8[0] = *(const v8bf*)(qrow + ks * 32 + half * 8);
      qf[ks].h8[1] = *(const v8bf*)(qrow + ks * 32 + 16 + half * 8);
    }
  }

  // constant all-ones B fragment: rowsum(P) = P @ ones  (broadcast to lanes)
  bfrag onesf;
#pragma unroll
  for (int i = 0; i < 16; ++i) onesf.v[i] = f2bf(1.0f);

  v8f o[4] = {};
  float mrow[8], lrow[8];
#pragma unroll
  for (int r = 0; r < 8; ++r) { mrow[r] = -1e30f; lrow[r] = 0.f; }

  const int ld_row = tid >> 2, ld_seg = (tid & 3) * 16;  // 64x64 / 256 = 16 each

  auto issue = [&](int kt, int buf) {
    const bf16_t* ks = qkv + (size_t)(b * S + kt * 64 + ld_row) * TD + D + h * DH + ld_seg;
    cp16(&Ksh[buf][ld_row][ld_seg], ks);
    cp16(&Ksh[buf][ld_row][ld_seg + 8], ks + 8);
    const bf16_t* vs = Vt + ((size_t)(b * 8 + h) * DH + ld_row) * S + kt * 64 + ld_seg;
    cp16(&Vsh[buf][ld_row][ld_seg], vs);
    cp16(&Vsh[buf][ld_row][ld_seg + 8], vs + 8);
  };

  const int ktend = (causal ? (q0 + 128) : S) >> 6;
  const float scl2 = 0.125f * 1.44269504088896f;  // log2(e)/sqrt(dh)

  issue(0, 0);

  for (int kt = 0; kt < ktend; ++kt) {
    const int cur = kt & 1;
    async_wait0();
    __syncthreads();
    if (kt + 1 < ktend) issue(kt + 1, cur ^ 1);

    // S = Q @ K^T (log2 domain scores)
    v8f sc[4];
#pragma unroll
    for (int nt = 0; nt < 4; ++nt) {
      v8f s = {};
#pragma unroll
      for (int ks = 0; ks < 2; ++ks) {
        bfrag kf;
        const bf16_t* kp = &Ksh[cur][nt * 16 + l16][ks * 32 + half * 8];
        kf.h8[0] = *(const v8bf*)(kp);
        kf.h8[1] = *(const v8bf*)(kp + 16);
        s = wmma_bf16(qf[ks], kf, s);
      }
      sc[nt] = s;
    }

    // scale (log2 domain); mask only on diagonal tiles (wave-uniform branch)
    const bool need_mask = causal && (kt * 64 + 63 > qw);
    float rmax[8];
    if (need_mask) {
#pragma unroll
      for (int r = 0; r < 8; ++r) {
        const int qrow = qw + r + half * 8;
        float mx = -1e30f;
#pragma unroll
        for (int nt = 0; nt < 4; ++nt) {
          const int kcol = kt * 64 + nt * 16 + l16;
          float v = sc[nt][r] * scl2;
          if (kcol > qrow) v = -1e30f;
          sc[nt][r] = v;
          mx = fmaxf(mx, v);
        }
        rmax[r] = mx;
      }
    } else {
#pragma unroll
      for (int r = 0; r < 8; ++r) {
        float mx = -1e30f;
#pragma unroll
        for (int nt = 0; nt < 4; ++nt) {
          const float v = sc[nt][r] * scl2;
          sc[nt][r] = v;
          mx = fmaxf(mx, v);
        }
        rmax[r] = mx;
      }
    }

    // online softmax update (exp2); row max across the 16-lane group on VALU
    float alpha[8];
#pragma unroll
    for (int r = 0; r < 8; ++r) {
      const float mnew = fmaxf(mrow[r], xor16_max(rmax[r]));
      alpha[r] = exp2f(mrow[r] - mnew);
      mrow[r] = mnew;
#pragma unroll
      for (int nt = 0; nt < 4; ++nt)
        sc[nt][r] = exp2f(sc[nt][r] - mnew);
#pragma unroll
      for (int dt = 0; dt < 4; ++dt) o[dt][r] *= alpha[r];
    }

    // P: C-layout f32 -> per-wave LDS bf16 -> A-fragment layout
#pragma unroll
    for (int nt = 0; nt < 4; ++nt)
#pragma unroll
      for (int r = 0; r < 8; ++r)
        Psh[wave][r + half * 8][nt * 16 + l16] = f2bf(sc[nt][r]);

    bfrag pf[2];
#pragma unroll
    for (int ks = 0; ks < 2; ++ks) {
      const bf16_t* pp = &Psh[wave][l16][ks * 32];
      pf[ks].h8[0] = *(const v8bf*)(pp + half * 8);
      pf[ks].h8[1] = *(const v8bf*)(pp + 16 + half * 8);
    }

    // row sums of P via WMMA against ones (every lane gets its row's sum)
    {
      v8f rs = {};
      rs = wmma_bf16(pf[0], onesf, rs);
      rs = wmma_bf16(pf[1], onesf, rs);
#pragma unroll
      for (int r = 0; r < 8; ++r)
        lrow[r] = lrow[r] * alpha[r] + rs[r];
    }

    // O += P @ V
#pragma unroll
    for (int dt = 0; dt < 4; ++dt) {
#pragma unroll
      for (int ks = 0; ks < 2; ++ks) {
        bfrag vf;
        const bf16_t* vp = &Vsh[cur][dt * 16 + l16][ks * 32 + half * 8];
        vf.h8[0] = *(const v8bf*)(vp);
        vf.h8[1] = *(const v8bf*)(vp + 16);
        o[dt] = wmma_bf16(pf[ks], vf, o[dt]);
      }
    }
  }

  // normalize, write O (bf16) as [B*S, D], head h at cols h*64..
#pragma unroll
  for (int dt = 0; dt < 4; ++dt)
#pragma unroll
    for (int r = 0; r < 8; ++r) {
      const int qrow = qw + r + half * 8;
      Oout[(size_t)(b * S + qrow) * D + h * DH + dt * 16 + l16] = f2bf(o[dt][r] / lrow[r]);
    }
}

// ---------------------------------------------------------------------------
extern "C" void kernel_launch(void* const* d_in, const int* in_sizes, int n_in,
                              void* d_out, int out_size, void* d_ws, size_t ws_size,
                              hipStream_t stream) {
  (void)in_sizes; (void)n_in; (void)out_size; (void)ws_size;
  const float* x      = (const float*)d_in[0];   // [2,4096,512]
  const float* W_in   = (const float*)d_in[1];   // [512,1536]
  const float* b_in   = (const float*)d_in[2];   // [1536]
  const float* W_out  = (const float*)d_in[3];   // [512,512]
  const float* b_out  = (const float*)d_in[4];   // [512]
  const int*   causal = (const int*)d_in[5];     // [1]
  float* out = (float*)d_out;                    // [2,4096,512]

  constexpr int M = 2 * 4096, D = 512, TD = 1536, S = 4096;

  char* w = (char*)d_ws;
  bf16_t* qkv   = (bf16_t*)w;                 w += (size_t)M * TD * 2;   // 25.2 MB
  bf16_t* attn  = (bf16_t*)w;                 w += (size_t)M * D * 2;    //  8.4 MB
  bf16_t* xb    = (bf16_t*)w;                 w += (size_t)M * D * 2;    //  8.4 MB
  bf16_t* WinT  = (bf16_t*)w;                 w += (size_t)TD * D * 2;   //  1.6 MB
  bf16_t* WoutT = (bf16_t*)w;                 w += (size_t)D * D * 2;    //  0.5 MB
  bf16_t* Vt    = (bf16_t*)w;                                            //  8.4 MB

  // prep: x -> bf16 ; W_in^T, W_out^T -> bf16 [N][K]
  cvt_bf16_kernel<<<(M * D) / (8 * 256), 256, 0, stream>>>(x, xb, M * D);
  transpose_cvt_kernel<<<dim3(TD / 64, D / 64), 256, 0, stream>>>(W_in, WinT, D, TD);
  transpose_cvt_kernel<<<dim3(D / 64, D / 64), 256, 0, stream>>>(W_out, WoutT, D, D);

  // 1) qkv = x @ W_in + b_in
  gemm_bias_kernel<bf16_t><<<dim3(M / 128, TD / 128), 256, 0, stream>>>(
      xb, WinT, b_in, qkv, M, TD, D);

  // V -> [B][H][dh][S]
  v_transpose_kernel<<<dim3(S / 64, 8, 2), 256, 0, stream>>>(qkv, Vt);

  // 2) flash attention
  flash_attn_kernel<<<dim3(S / 128, 8, 2), 256, 0, stream>>>(qkv, Vt, attn, causal);

  // 3) out = attn @ W_out + b_out
  gemm_bias_kernel<float><<<dim3(M / 128, D / 128), 256, 0, stream>>>(
      attn, WoutT, b_out, out, M, D, D);
}
